// DynamicConv1dTBC_14955076124989
// MI455X (gfx1250) — matrix operand order repaired and checked by reading, not compiled, not run on hardware
//
#include <hip/hip_runtime.h>

#define T_DIM   2048
#define B_DIM   4
#define C_DIM   1024
#define H_DIM   16
#define K_TAPS  31
#define PAD_LF  15
#define F_DIM   (H_DIM * K_TAPS)     // 496
#define F_PAD   512                  // padded to 32 N-tiles -> branch-free WMMA loop
#define M_ROWS  (T_DIM * B_DIM)      // 8192

typedef __attribute__((ext_vector_type(16))) _Float16 v16h;
typedef __attribute__((ext_vector_type(8)))  _Float16 v8h;
typedef __attribute__((ext_vector_type(4)))  _Float16 v4h;
typedef __attribute__((ext_vector_type(8)))  float    v8f;

#define AS_STRIDE 1032   // halves per LDS A row (1024 + 8 pad -> stride%64dw==4: conflict-free b128)
#define LS_STRIDE 520    // floats per LDS logits row (512 used -> pad 520)

union SharedU {
    _Float16 a[16 * AS_STRIDE];   // 33,024 B : f16 staged x tile
    float    l[16 * LS_STRIDE];   // 33,280 B : f32 logits / softmaxed weights
};

static __device__ __forceinline__ v16h load_a_frag(const _Float16* arow, int kc) {
    v8h alo = *(const v8h*)(arow + kc);
    v8h ahi = *(const v8h*)(arow + kc + 8);
    return __builtin_shufflevector(alo, ahi,
                                   0, 1, 2, 3, 4, 5, 6, 7,
                                   8, 9, 10, 11, 12, 13, 14, 15);
}

// ---------------------------------------------------------------------------
// Kernel 0: convert Ww (f32, [496,1024]) -> f16 [512,1024] in workspace,
// zero-padding rows 496..511 so the GEMM loop needs no tail guard.
// ---------------------------------------------------------------------------
__global__ __launch_bounds__(256) void cvt_w_kernel(const float* __restrict__ Ww,
                                                    _Float16* __restrict__ Wwh) {
    int i = (blockIdx.x * 256 + threadIdx.x) * 4;   // 0 .. F_PAD*C_DIM-4
    v4h h = { (_Float16)0.f, (_Float16)0.f, (_Float16)0.f, (_Float16)0.f };
    if (i < F_DIM * C_DIM) {
        float4 v = *(const float4*)(Ww + i);
        h = (v4h){ (_Float16)v.x, (_Float16)v.y, (_Float16)v.z, (_Float16)v.w };
    }
    *(v4h*)(Wwh + i) = h;
}

// ---------------------------------------------------------------------------
// Kernel 1: fused  GEMM (WMMA f16->f32)  +  softmax(K)  +  dynamic depthwise conv
// One block = 16 flattened (t,b) rows = one M-tile. 256 threads = 8 waves.
// Each wave owns N-tiles {w, w+8, w+16, w+24} of 32 (branch-free).
// GEMM K-loop is software-pipelined (2 stages, ping-pong via unroll 2);
// waves_per_eu(4) caps VGPRs at 256 to preserve occupancy for the
// memory-latency-bound conv phase.
// ---------------------------------------------------------------------------
__global__ __launch_bounds__(256)
__attribute__((amdgpu_waves_per_eu(4)))
void dynconv_fused_kernel(const float* __restrict__ x,
                          const _Float16* __restrict__ Wwh,
                          float* __restrict__ out) {
    __shared__ SharedU smem;

    const int tid   = threadIdx.x;
    const int mtile = blockIdx.x;          // 0..511
    const int g0    = mtile * 16;          // first flat row (t*B+b)

    // ---- Phase 1: stage x tile [16 x 1024] f32 -> f16 LDS ----
    #pragma unroll
    for (int i = 0; i < 16; ++i) {
        int idx = i * 256 + tid;           // 4096 float4 slots
        int row = idx >> 8;                // 256 float4 per row
        int c4  = (idx & 255) << 2;
        float4 v = *(const float4*)(x + (size_t)(g0 + row) * C_DIM + c4);
        v4h h = { (_Float16)v.x, (_Float16)v.y, (_Float16)v.z, (_Float16)v.w };
        *(v4h*)(&smem.a[row * AS_STRIDE + c4]) = h;
    }
    __syncthreads();

    // ---- Phase 2: WMMA GEMM: logits[16 x 512] = A(16x1024) * Wwh^T ----
    const int lane = tid & 31;
    const int wid  = tid >> 5;             // 0..7
    const int am   = lane & 15;            // A: M row
    const int akh  = lane >> 4;            // A: K half select
    const int bn   = lane & 15;            // B: N col within tile
    const int bkh  = lane >> 4;            // B: K half select

    const _Float16* arow = &smem.a[am * AS_STRIDE + akh * 16];
    const _Float16* bp0  = Wwh + (size_t)((wid +  0) * 16 + bn) * C_DIM + bkh * 16;
    const _Float16* bp1  = Wwh + (size_t)((wid +  8) * 16 + bn) * C_DIM + bkh * 16;
    const _Float16* bp2  = Wwh + (size_t)((wid + 16) * 16 + bn) * C_DIM + bkh * 16;
    const _Float16* bp3  = Wwh + (size_t)((wid + 24) * 16 + bn) * C_DIM + bkh * 16;

    v8f acc0 = {}, acc1 = {}, acc2 = {}, acc3 = {};

    // Pipeline prologue: stage 0
    v16h a_c  = load_a_frag(arow, 0);
    v16h b0_c = *(const v16h*)(bp0);
    v16h b1_c = *(const v16h*)(bp1);
    v16h b2_c = *(const v16h*)(bp2);
    v16h b3_c = *(const v16h*)(bp3);

    #pragma clang loop unroll_count(2)
    for (int kc = 0; kc < C_DIM - 32; kc += 32) {
        const int kn = kc + 32;
        // Issue next-stage loads first (stay outstanding across the WMMAs)
        v16h a_n  = load_a_frag(arow, kn);
        v16h b0_n = *(const v16h*)(bp0 + kn);
        v16h b1_n = *(const v16h*)(bp1 + kn);
        v16h b2_n = *(const v16h*)(bp2 + kn);
        v16h b3_n = *(const v16h*)(bp3 + kn);
        // Compute on current stage
        acc0 = __builtin_amdgcn_wmma_f32_16x16x32_f16(false, a_c, false, b0_c,
                                                      (short)0, acc0, false, false);
        acc1 = __builtin_amdgcn_wmma_f32_16x16x32_f16(false, a_c, false, b1_c,
                                                      (short)0, acc1, false, false);
        acc2 = __builtin_amdgcn_wmma_f32_16x16x32_f16(false, a_c, false, b2_c,
                                                      (short)0, acc2, false, false);
        acc3 = __builtin_amdgcn_wmma_f32_16x16x32_f16(false, a_c, false, b3_c,
                                                      (short)0, acc3, false, false);
        // Rotate (unroll 2 -> ping-pong, no movs)
        a_c = a_n; b0_c = b0_n; b1_c = b1_n; b2_c = b2_n; b3_c = b3_n;
    }
    // Pipeline epilogue: last stage
    acc0 = __builtin_amdgcn_wmma_f32_16x16x32_f16(false, a_c, false, b0_c,
                                                  (short)0, acc0, false, false);
    acc1 = __builtin_amdgcn_wmma_f32_16x16x32_f16(false, a_c, false, b1_c,
                                                  (short)0, acc1, false, false);
    acc2 = __builtin_amdgcn_wmma_f32_16x16x32_f16(false, a_c, false, b2_c,
                                                  (short)0, acc2, false, false);
    acc3 = __builtin_amdgcn_wmma_f32_16x16x32_f16(false, a_c, false, b3_c,
                                                  (short)0, acc3, false, false);

    __syncthreads();          // done reading smem.a; safe to alias as smem.l

    // ---- Phase 3: spill accumulators to LDS logits ----
    {
        const int mh = (lane >> 4) * 8;    // D layout: VGPR v holds M = v + 8*(lane>=16)
        const int nl = lane & 15;
        #pragma unroll
        for (int v = 0; v < 8; ++v) smem.l[(mh + v) * LS_STRIDE + (wid +  0) * 16 + nl] = acc0[v];
        #pragma unroll
        for (int v = 0; v < 8; ++v) smem.l[(mh + v) * LS_STRIDE + (wid +  8) * 16 + nl] = acc1[v];
        #pragma unroll
        for (int v = 0; v < 8; ++v) smem.l[(mh + v) * LS_STRIDE + (wid + 16) * 16 + nl] = acc2[v];
        #pragma unroll
        for (int v = 0; v < 8; ++v) smem.l[(mh + v) * LS_STRIDE + (wid + 24) * 16 + nl] = acc3[v];
    }
    __syncthreads();

    // ---- Phase 4: softmax over K taps, in place (16 rows x 16 heads) ----
    // Padded columns 496..511 hold garbage but are never read (h*31+k <= 495).
    {
        const int m = tid >> 4;
        const int h = tid & 15;
        float* p = &smem.l[m * LS_STRIDE + h * K_TAPS];
        float mx = p[0];
        #pragma unroll
        for (int k = 1; k < K_TAPS; ++k) mx = fmaxf(mx, p[k]);
        float s = 0.0f;
        #pragma unroll
        for (int k = 0; k < K_TAPS; ++k) {
            float e = __expf(p[k] - mx);
            s += e;
            p[k] = e;
        }
        float inv = 1.0f / s;
        #pragma unroll
        for (int k = 0; k < K_TAPS; ++k) p[k] *= inv;
    }
    __syncthreads();

    // ---- Phase 5: depthwise dynamic conv ----
    for (int it = 0; it < 16; ++it) {
        int idx = it * 256 + tid;          // 4096 float4 output slots
        int m   = idx >> 8;
        int c0  = (idx & 255) << 2;
        int h   = c0 >> 6;                 // R = 64 channels/head
        int gr  = g0 + m;
        int t   = gr >> 2;                 // B = 4
        int b   = gr & 3;
        const float* wrow = &smem.l[m * LS_STRIDE + h * K_TAPS];
        float4 av = {0.0f, 0.0f, 0.0f, 0.0f};
        #pragma unroll
        for (int k = 0; k < K_TAPS; ++k) {
            int tt = t + k - PAD_LF;
            if (tt >= 0 && tt < T_DIM) {
                float4 xv = *(const float4*)(x + ((size_t)tt * B_DIM + b) * C_DIM + c0);
                float wk = wrow[k];
                av.x = fmaf(wk, xv.x, av.x);
                av.y = fmaf(wk, xv.y, av.y);
                av.z = fmaf(wk, xv.z, av.z);
                av.w = fmaf(wk, xv.w, av.w);
            }
        }
        *(float4*)(out + (size_t)gr * C_DIM + c0) = av;
    }
}

// ---------------------------------------------------------------------------
extern "C" void kernel_launch(void* const* d_in, const int* in_sizes, int n_in,
                              void* d_out, int out_size, void* d_ws, size_t ws_size,
                              hipStream_t stream) {
    (void)in_sizes; (void)n_in; (void)out_size; (void)ws_size;
    const float* x  = (const float*)d_in[0];   // [T,B,C] f32
    const float* Ww = (const float*)d_in[1];   // [H*K, C] f32
    float* out      = (float*)d_out;           // [T,B,C] f32
    _Float16* Wwh   = (_Float16*)d_ws;         // 512*1024 f16 (1 MB, zero-padded)

    cvt_w_kernel<<<(F_PAD * C_DIM / 4) / 256, 256, 0, stream>>>(Ww, Wwh);
    dynconv_fused_kernel<<<M_ROWS / 16, 256, 0, stream>>>(x, Wwh, out);
}
